// LinearAttentionRelation_14087492731622
// MI455X (gfx1250) — matrix operand order repaired and checked by reading, not compiled, not run on hardware
//
#include <hip/hip_runtime.h>
#include <hip/hip_bf16.h>

// ---------------------------------------------------------------------------
// Linear attention (heads=8, d=64) + Wo + residual + BatchNorm for MI455X
// (gfx1250, wave32, WMMA bf16 + async LDS copies).
//
// Pipeline (all on `stream`):
//   0) zero ctx/stat scratch
//   1) convert Wq/Wk/Wv/Wo and x to bf16 (hoists all cvts out of GEMM loops)
//   2) K1: async-stage x tile to LDS; Q/K/V projections (WMMA), softmax(K)
//      over head-dim, ctx += softmax(k) @ v^T (WMMA + f32 atomics)
//   3) K2: async-stage q tile; attn = ctx^T @ q, out = Wo @ attn + residual,
//      per-channel BN statistics
//   4) K3: in-place BN normalize of d_out
//
// Workspace (~138 MB): x bf16 (64MB) | Q scratch bf16 (64MB) | weights bf16
// (2MB) | ctx f32 (2MB) | gsum/gsumsq (4KB)
// ---------------------------------------------------------------------------

#define C_DIM   512
#define HEADS   8
#define HDIM    64
#define NPOS    4096          // H*W
#define BATCH   16
#define POSWG   64            // positions per workgroup
#define NWAVES  16
#define BN_CNT  65536.0f      // B*H*W
#define BN_EPS  1e-5f

typedef __attribute__((ext_vector_type(8)))  float        v8f;
typedef __attribute__((ext_vector_type(16))) __bf16       v16bf;
typedef __attribute__((ext_vector_type(2)))  __bf16       v2bf;
typedef __attribute__((ext_vector_type(4)))  unsigned int v4u;
typedef __attribute__((ext_vector_type(8)))  unsigned int v8u;

static __device__ __forceinline__ v8f wmma_bf16(v16bf a, v16bf b, v8f c) {
  return __builtin_amdgcn_wmma_f32_16x16x32_bf16(
      /*neg_a=*/false, a, /*neg_b=*/false, b,
      /*c_mod=*/(short)0, c, /*reuse_a=*/false, /*reuse_b=*/false);
}

// Async global->LDS 16-byte copy (CDNA5 GLOBAL_LOAD_ASYNC_TO_LDS_B128,
// tracked by ASYNCcnt). lds_off is the LDS byte offset (low 32 bits of the
// flat shared-aperture address, per ISA 10.2); gaddr is a global pointer.
static __device__ __forceinline__ void async_copy_b128(unsigned lds_off,
                                                       const void* gaddr) {
  asm volatile("global_load_async_to_lds_b128 %0, %1, off"
               :: "v"(lds_off), "v"(gaddr)
               : "memory");
}
static __device__ __forceinline__ void wait_async0() {
  asm volatile("s_wait_asynccnt 0x0" ::: "memory");
}

// A fragment (16x32 bf16). `row` points at 32 consecutive bf16 K-values of
// this lane's M-row. Lane-group g needs K pairs {g*8..} and {16+g*8..} ==
// dwords [g*4..g*4+3] and [8+g*4..8+g*4+3] == two 16-byte loads.
static __device__ __forceinline__ v16bf load_afrag(const __bf16* row, int g) {
  const v4u* p = (const v4u*)row;
  v4u a0 = p[g];
  v4u a1 = p[2 + g];
  v8u u = __builtin_shufflevector(a0, a1, 0, 1, 2, 3, 4, 5, 6, 7);
  return __builtin_bit_cast(v16bf, u);
}

// B fragment (32x16 bf16): p points at K-row 0 of the step (row stride 64);
// lane supplies column N=m, K in table order {g*8..+7, 16+g*8..+7}.
static __device__ __forceinline__ v16bf load_bfrag(const __bf16* p16, int g, int m) {
  v16bf r;
#pragma unroll
  for (int s = 0; s < 16; ++s) {
    int k = (s < 8) ? (g * 8 + s) : (16 + g * 8 + (s - 8));
    r[s] = p16[k * 64 + m];
  }
  return r;
}

// 64(M) x 32(N) x 512(K) projection tile: bf16 weights (global, L2-resident)
// x bf16 activations staged in LDS ([512][64] row-major).
static __device__ __forceinline__ void gemm_proj(const __bf16* __restrict__ W,
                                                 const __bf16* xt,
                                                 int og, int phOff, int m,
                                                 int g, int lane,
                                                 v8f (&acc)[4][2]) {
  for (int ks = 0; ks < 16; ++ks) {
    v16bf aF[4];
#pragma unroll
    for (int mt = 0; mt < 4; ++mt)
      aF[mt] = load_afrag(W + (size_t)(og * 64 + mt * 16 + m) * C_DIM + ks * 32, g);
    v16bf bF[2];
#pragma unroll
    for (int nt = 0; nt < 2; ++nt)
      bF[nt] = load_bfrag(xt + (ks * 32) * 64 + phOff + nt * 16, g, m);
#pragma unroll
    for (int mt = 0; mt < 4; ++mt)
#pragma unroll
      for (int nt = 0; nt < 2; ++nt)
        acc[mt][nt] = wmma_bf16(aF[mt], bF[nt], acc[mt][nt]);
  }
}

// ---------------------------------------------------------------------------
// Kernel 1: Q/K/V projections, softmax(K) over head-dim, ctx += k @ v^T
// Grid (NPOS/POSWG, BATCH), block 512 (16 waves).
// Wave w: head og=w/2 (channels 64*og..+63), position half ph=w&1.
// ---------------------------------------------------------------------------
__global__ __launch_bounds__(512) void k_qkv_ctx(
    const __bf16* __restrict__ xbf,
    const __bf16* __restrict__ Wq, const __bf16* __restrict__ Wk,
    const __bf16* __restrict__ Wv,
    __bf16* __restrict__ qscr, float* __restrict__ gctx) {
  __shared__ __bf16 xtile[C_DIM * POSWG];     // [c][pos]   x tile (64 KB)
  __shared__ __bf16 kbuf[NWAVES][HDIM * 32];  // [d][pos]   A-operand order
  __shared__ __bf16 vbuf[NWAVES][32 * HDIM];  // [pos][e]   B-operand order

  const int pt   = blockIdx.x;
  const int b    = blockIdx.y;
  const int lane = (int)(threadIdx.x & 31);
  const int w    = (int)(threadIdx.x >> 5);
  const int og   = w >> 1;
  const int ph   = w & 1;
  const int m    = lane & 15;
  const int g    = lane >> 4;
  const int phOff = ph * 32;

  const v8f vzero = {0.f, 0.f, 0.f, 0.f, 0.f, 0.f, 0.f, 0.f};

  // ---- async-stage x tile [512 x 64] into LDS (4096 x b128) ----
  {
    const char* gsrc =
        (const char*)(xbf + (size_t)b * C_DIM * NPOS + (size_t)pt * POSWG);
    const unsigned ldsbase = (unsigned)(uintptr_t)&xtile[0];
    for (int i = (int)threadIdx.x; i < 4096; i += 512) {
      int c = i >> 3, ch = i & 7;
      async_copy_b128(ldsbase + (unsigned)i * 16,
                      gsrc + (size_t)c * (NPOS * 2) + ch * 16);
    }
    wait_async0();
    __syncthreads();
  }

  // ---- Q projection -> bf16 scratch ----
  {
    v8f acc[4][2];
#pragma unroll
    for (int mt = 0; mt < 4; ++mt)
#pragma unroll
      for (int nt = 0; nt < 2; ++nt) acc[mt][nt] = vzero;
    gemm_proj(Wq, xtile, og, phOff, m, g, lane, acc);
#pragma unroll
    for (int mt = 0; mt < 4; ++mt)
#pragma unroll
      for (int nt = 0; nt < 2; ++nt)
#pragma unroll
        for (int r = 0; r < 8; ++r) {
          int c   = og * 64 + mt * 16 + g * 8 + r;
          int pos = pt * POSWG + phOff + nt * 16 + m;
          qscr[((size_t)b * C_DIM + c) * NPOS + pos] = (__bf16)acc[mt][nt][r];
        }
  }

  // ---- V projection -> LDS slice in B-operand order [pos][e] ----
  {
    v8f acc[4][2];
#pragma unroll
    for (int mt = 0; mt < 4; ++mt)
#pragma unroll
      for (int nt = 0; nt < 2; ++nt) acc[mt][nt] = vzero;
    gemm_proj(Wv, xtile, og, phOff, m, g, lane, acc);
#pragma unroll
    for (int mt = 0; mt < 4; ++mt)
#pragma unroll
      for (int nt = 0; nt < 2; ++nt)
#pragma unroll
        for (int rr = 0; rr < 4; ++rr) {
          v2bf pk;
          pk[0] = (__bf16)acc[mt][nt][2 * rr];
          pk[1] = (__bf16)acc[mt][nt][2 * rr + 1];
          *(v2bf*)&vbuf[w][(nt * 16 + m) * 64 + mt * 16 + g * 8 + 2 * rr] = pk;
        }
  }

  // ---- K projection + softmax over head-dim (wave-internal) -> LDS ----
  {
    v8f acc[4][2];
#pragma unroll
    for (int mt = 0; mt < 4; ++mt)
#pragma unroll
      for (int nt = 0; nt < 2; ++nt) acc[mt][nt] = vzero;
    gemm_proj(Wk, xtile, og, phOff, m, g, lane, acc);
    // One position column's 64 head-dim values live in this lane's 32
    // accumulator slots plus its partner lane (L ^ 16).
#pragma unroll
    for (int nt = 0; nt < 2; ++nt) {
      float mx = -3.4e38f;
#pragma unroll
      for (int mt = 0; mt < 4; ++mt)
#pragma unroll
        for (int r = 0; r < 8; ++r) mx = fmaxf(mx, acc[mt][nt][r]);
      mx = fmaxf(mx, __shfl_xor(mx, 16, 32));
      float s = 0.f;
#pragma unroll
      for (int mt = 0; mt < 4; ++mt)
#pragma unroll
        for (int r = 0; r < 8; ++r) {
          float e = expf(acc[mt][nt][r] - mx);
          acc[mt][nt][r] = e;
          s += e;
        }
      s += __shfl_xor(s, 16, 32);
      float inv = 1.f / s;
#pragma unroll
      for (int mt = 0; mt < 4; ++mt)
#pragma unroll
        for (int r = 0; r < 8; ++r)
          kbuf[w][(mt * 16 + g * 8 + r) * 32 + nt * 16 + m] =
              (__bf16)(acc[mt][nt][r] * inv);
    }
  }

  // ---- ctx[d,e] += sum_pos k[d,pos] * v[e,pos]  (K = 32 positions) ----
  float* ctxh = gctx + ((size_t)b * HEADS + og) * HDIM * HDIM;
  for (int mt = 0; mt < 4; ++mt) {
    v16bf aF = load_afrag(&kbuf[w][(mt * 16 + m) * 32], g);
    for (int et = 0; et < 4; ++et) {
      v16bf bF = load_bfrag(&vbuf[w][et * 16], g, m);
      v8f cc = wmma_bf16(aF, bF, vzero);
#pragma unroll
      for (int r = 0; r < 8; ++r) {
        int d = mt * 16 + g * 8 + r;
        int e = et * 16 + m;
        atomicAdd(&ctxh[(size_t)d * HDIM + e], cc[r]);
      }
    }
  }
}

// ---------------------------------------------------------------------------
// Kernel 2: attn = ctx^T @ q (per head), out = Wo @ attn + x -> d_out,
// plus per-channel sum / sum^2 accumulation for BatchNorm.
// ---------------------------------------------------------------------------
__global__ __launch_bounds__(512) void k_attn_out(
    const float* __restrict__ x, const __bf16* __restrict__ Wo,
    const __bf16* __restrict__ qscr, const float* __restrict__ gctx,
    float* __restrict__ y, float* __restrict__ gsum,
    float* __restrict__ gsumsq) {
  __shared__ __bf16 qs[C_DIM * POSWG];  // q tile; reused for attn tile

  const int pt   = blockIdx.x;
  const int b    = blockIdx.y;
  const int lane = (int)(threadIdx.x & 31);
  const int w    = (int)(threadIdx.x >> 5);
  const int og   = w >> 1;
  const int ph   = w & 1;
  const int m    = lane & 15;
  const int g    = lane >> 4;

  const v8f vzero = {0.f, 0.f, 0.f, 0.f, 0.f, 0.f, 0.f, 0.f};

  // ---- async-stage q tile [512 x 64] into LDS ----
  {
    const char* gsrc =
        (const char*)(qscr + (size_t)b * C_DIM * NPOS + (size_t)pt * POSWG);
    const unsigned ldsbase = (unsigned)(uintptr_t)&qs[0];
    for (int i = (int)threadIdx.x; i < 4096; i += 512) {
      int c = i >> 3, ch = i & 7;
      async_copy_b128(ldsbase + (unsigned)i * 16,
                      gsrc + (size_t)c * (NPOS * 2) + ch * 16);
    }
    wait_async0();
    __syncthreads();
  }

  // attn[e,pos] = sum_d ctx[d,e] * q[d,pos]   (K = 64, two WMMA k-steps)
  v8f acc[4][2];
#pragma unroll
  for (int et = 0; et < 4; ++et)
#pragma unroll
    for (int nt = 0; nt < 2; ++nt) acc[et][nt] = vzero;

  const float* ctxh = gctx + ((size_t)b * HEADS + og) * HDIM * HDIM;
#pragma unroll
  for (int ks = 0; ks < 2; ++ks) {
    v16bf aF[4];
#pragma unroll
    for (int et = 0; et < 4; ++et) {
#pragma unroll
      for (int v = 0; v < 8; ++v) {
        int kk = ((v & 4) ? 16 : 0) + (g << 3) + ((v & 3) << 1);
        int d  = ks * 32 + kk;
        aF[et][2 * v]     = (__bf16)ctxh[(size_t)d * HDIM + et * 16 + m];
        aF[et][2 * v + 1] = (__bf16)ctxh[(size_t)(d + 1) * HDIM + et * 16 + m];
      }
    }
    v16bf bF[2];
#pragma unroll
    for (int nt = 0; nt < 2; ++nt)
      bF[nt] = load_bfrag(&qs[(og * 64 + ks * 32) * 64 + ph * 32 + nt * 16], g, m);
#pragma unroll
    for (int et = 0; et < 4; ++et)
#pragma unroll
      for (int nt = 0; nt < 2; ++nt)
        acc[et][nt] = wmma_bf16(aF[et], bF[nt], acc[et][nt]);
  }

  // Write attn back into qs (each wave touches only its own head rows and
  // position half, which only it has read) then sync before the Wo GEMM.
#pragma unroll
  for (int et = 0; et < 4; ++et)
#pragma unroll
    for (int nt = 0; nt < 2; ++nt)
#pragma unroll
      for (int r = 0; r < 8; ++r) {
        int c = og * 64 + et * 16 + g * 8 + r;
        int p = ph * 32 + nt * 16 + m;
        qs[c * 64 + p] = (__bf16)acc[et][nt][r];
      }
  __syncthreads();

  // out = Wo @ attn  (K = 512)
  v8f oacc[4][2];
#pragma unroll
  for (int mt = 0; mt < 4; ++mt)
#pragma unroll
    for (int nt = 0; nt < 2; ++nt) oacc[mt][nt] = vzero;
  for (int ks = 0; ks < 16; ++ks) {
    v16bf aF[4];
#pragma unroll
    for (int mt = 0; mt < 4; ++mt)
      aF[mt] = load_afrag(Wo + (size_t)(og * 64 + mt * 16 + m) * C_DIM + ks * 32, g);
    v16bf bF[2];
#pragma unroll
    for (int nt = 0; nt < 2; ++nt)
      bF[nt] = load_bfrag(&qs[(ks * 32) * 64 + ph * 32 + nt * 16], g, m);
#pragma unroll
    for (int mt = 0; mt < 4; ++mt)
#pragma unroll
      for (int nt = 0; nt < 2; ++nt)
        oacc[mt][nt] = wmma_bf16(aF[mt], bF[nt], oacc[mt][nt]);
  }

  // Epilogue: residual add (f32 x), store y, accumulate BN statistics.
  const int posW = pt * POSWG + ph * 32;
#pragma unroll
  for (int mt = 0; mt < 4; ++mt) {
#pragma unroll
    for (int r = 0; r < 8; ++r) {
      int c = og * 64 + mt * 16 + g * 8 + r;
      float s = 0.f, ss = 0.f;
#pragma unroll
      for (int nt = 0; nt < 2; ++nt) {
        int pos = posW + nt * 16 + m;
        size_t idx = ((size_t)b * C_DIM + c) * NPOS + pos;
        float yv = oacc[mt][nt][r] + x[idx];
        y[idx] = yv;
        s += yv;
        ss += yv * yv;
      }
      // butterfly over the 16 lanes of this half-wave (channel is fixed)
#pragma unroll
      for (int mask = 8; mask >= 1; mask >>= 1) {
        s  += __shfl_xor(s, mask, 32);
        ss += __shfl_xor(ss, mask, 32);
      }
      if (m == 0) {
        atomicAdd(&gsum[c], s);
        atomicAdd(&gsumsq[c], ss);
      }
    }
  }
}

// ---------------------------------------------------------------------------
// Kernel 3: in-place BatchNorm normalize of d_out with batch statistics.
// ---------------------------------------------------------------------------
__global__ void k_bn(const float* __restrict__ gsum,
                     const float* __restrict__ gsumsq,
                     const float* __restrict__ gamma,
                     const float* __restrict__ beta,
                     float* __restrict__ y, int total) {
  int idx = (int)(blockIdx.x * blockDim.x + threadIdx.x);
  if (idx >= total) return;
  int c = (idx >> 12) & (C_DIM - 1);
  float mean = gsum[c] * (1.0f / BN_CNT);
  float var  = gsumsq[c] * (1.0f / BN_CNT) - mean * mean;
  float inv  = rsqrtf(var + BN_EPS) * gamma[c];
  y[idx] = (y[idx] - mean) * inv + beta[c];
}

// f32 -> bf16 conversion, 2 elements/thread (packed dword stores).
__global__ void k_cvt_bf16(const float* __restrict__ src,
                           __bf16* __restrict__ dst, int n2) {
  int i = (int)(blockIdx.x * blockDim.x + threadIdx.x);
  if (i >= n2) return;
  float2 f = ((const float2*)src)[i];
  v2bf p;
  p[0] = (__bf16)f.x;
  p[1] = (__bf16)f.y;
  ((v2bf*)dst)[i] = p;
}

__global__ void k_zero(float* __restrict__ p, int n) {
  int i = (int)(blockIdx.x * blockDim.x + threadIdx.x);
  if (i < n) p[i] = 0.f;
}

// ---------------------------------------------------------------------------
extern "C" void kernel_launch(void* const* d_in, const int* in_sizes, int n_in,
                              void* d_out, int out_size, void* d_ws,
                              size_t ws_size, hipStream_t stream) {
  const float* x     = (const float*)d_in[0];
  const float* Wq    = (const float*)d_in[1];
  const float* Wk    = (const float*)d_in[2];
  const float* Wv    = (const float*)d_in[3];
  const float* Wo    = (const float*)d_in[4];
  const float* gamma = (const float*)d_in[5];
  const float* beta  = (const float*)d_in[6];
  float* out = (float*)d_out;

  const size_t xElems = (size_t)BATCH * C_DIM * NPOS;  // 33554432
  const int    wElems = C_DIM * C_DIM;                 // 262144
  const int    ctxElems = BATCH * HEADS * HDIM * HDIM; // 524288

  __bf16* xbf   = (__bf16*)d_ws;
  __bf16* qscr  = xbf + xElems;
  __bf16* wbf   = qscr + xElems;                 // 4 matrices back-to-back
  float*  gctx  = (float*)(wbf + 4 * wElems);
  float*  gsum  = gctx + ctxElems;
  float*  gsumsq = gsum + C_DIM;

  // 0) zero ctx + stats (workspace is poisoned; re-accumulated every call)
  {
    int n = ctxElems + 2 * C_DIM;
    k_zero<<<(n + 255) / 256, 256, 0, stream>>>(gctx, n);
  }
  // 1) bf16 conversions (weights then x); L2-resident afterwards
  {
    int wn2 = wElems / 2;
    k_cvt_bf16<<<(wn2 + 255) / 256, 256, 0, stream>>>(Wq, wbf + 0 * wElems, wn2);
    k_cvt_bf16<<<(wn2 + 255) / 256, 256, 0, stream>>>(Wk, wbf + 1 * wElems, wn2);
    k_cvt_bf16<<<(wn2 + 255) / 256, 256, 0, stream>>>(Wv, wbf + 2 * wElems, wn2);
    k_cvt_bf16<<<(wn2 + 255) / 256, 256, 0, stream>>>(Wo, wbf + 3 * wElems, wn2);
    int xn2 = (int)(xElems / 2);
    k_cvt_bf16<<<(xn2 + 255) / 256, 256, 0, stream>>>(x, xbf, xn2);
  }
  // 2) projections + softmax + context
  k_qkv_ctx<<<dim3(NPOS / POSWG, BATCH), 512, 0, stream>>>(
      xbf, wbf + 0 * wElems, wbf + 1 * wElems, wbf + 2 * wElems, qscr, gctx);
  // 3) attention output + Wo + residual + BN stats
  k_attn_out<<<dim3(NPOS / POSWG, BATCH), 512, 0, stream>>>(
      x, wbf + 3 * wElems, qscr, gctx, out, gsum, gsumsq);
  // 4) normalize in place
  {
    int total = (int)(BATCH * C_DIM * NPOS);
    k_bn<<<(total + 255) / 256, 256, 0, stream>>>(gsum, gsumsq, gamma, beta,
                                                  out, total);
  }
}
